// TerrellPenroseSNN_81664508166569
// MI455X (gfx1250) — matrix operand order repaired and checked by reading, not compiled, not run on hardware
//
#include <hip/hip_runtime.h>
#include <hip/hip_bf16.h>
#include <cstdint>

// ---------------------------------------------------------------------------
// TerrellPenroseSNN on MI455X (gfx1250, wave32, WMMA + TDM)
//
// B=1024, T=100, IN=256, HID=1024, OUT=256.
// GEMMs: ~107 GFLOP bf16 (compute-bound); all per-step tensors are L2
// resident (192 MB L2, 23.3 TB/s HBM). Spikes are exactly {0,1} -> bf16
// activations are lossless; weights cast to bf16 once.
// Training-mode BN subtracts the batch mean, so fc1_b/fc2_b cancel exactly.
//
// GEMM structure (per workgroup, 8 waves):
//   - block tile 128(M) x 128(N); wave tile 32x64 = 2x4 WMMA frags
//   - weight tile (reused operand) staged into LDS by the Tensor Data Mover
//     (tensor_load_to_lds), double buffered, K-chunk = 64; TDM pad feature
//     gives a 160B LDS row pitch to spread banks
//   - spikes (A) read directly from L2 as 16B loads (+ global_prefetch_b8)
//   - v_wmma_f32_16x16x32_bf16, f32 accumulate, 8 WMMAs per K-step
//   - fused BN-stats epilogue: per-column sum/sumsq reduced via ds_add_f32
//     in LDS, then one global atomicAdd per column per block. This removes
//     the separate stats kernels (~5 MB/step of redundant L2 reads).
// ---------------------------------------------------------------------------

#define BDIM 1024
#define TDIM 100
#define INDIM 256
#define HIDDIM 1024
#define OUTDIM 256
#define BN_EPS 1e-5f

// GEMM tiling
#define TP_KC     64                       // K elements per LDS chunk
#define TP_BROWS  128                      // weight rows (=N cols) per block
#define TP_ROWB   160                      // LDS row pitch: 128B data + 32B pad
#define TP_BUFB   (TP_BROWS * TP_ROWB)     // 20480 B per buffer

typedef __attribute__((ext_vector_type(16))) __bf16 v16bf;
typedef __attribute__((ext_vector_type(8)))  __bf16 v8bf;
typedef __attribute__((ext_vector_type(8)))  float  v8f;

#if defined(__has_builtin)
#if __has_builtin(__builtin_amdgcn_tensor_load_to_lds)
#define TP_HAVE_TDM 1
#endif
#endif

// ---------------------------------------------------------------------------
// Zero helper
// ---------------------------------------------------------------------------
__global__ void tp_zero_kernel(float* __restrict__ p, int n) {
    int i = blockIdx.x * blockDim.x + threadIdx.x;
    if (i < n) p[i] = 0.0f;
}

// ---------------------------------------------------------------------------
// One-time prep: relativistic delay factors + weight conversion to bf16.
// ---------------------------------------------------------------------------
__global__ void tp_prep_kernel(const float* __restrict__ w1f,  // [HID x IN]
                               const float* __restrict__ w2f,  // [OUT x HID]
                               const float* __restrict__ cd,   // [1024]
                               const float* __restrict__ lf,   // [1]
                               __bf16* __restrict__ w1,
                               __bf16* __restrict__ w2,
                               float* __restrict__ delays) {
    int i = blockIdx.x * blockDim.x + threadIdx.x;
    if (i < HIDDIM * INDIM)  w1[i] = (__bf16)w1f[i];
    if (i < OUTDIM * HIDDIM) w2[i] = (__bf16)w2f[i];
    if (i < 1024) {
        float v = fminf(fmaxf(lf[0], 0.0f), 0.99f);
        float gamma = rsqrtf(1.0f - v * v);
        delays[i] = 1.0f / (1.0f + gamma * fabsf(cd[i]) * v);
    }
}

// ---------------------------------------------------------------------------
// Input membrane update + hard-threshold spike (forward only).
// Also zeroes the layer-1 BN stats accumulators for this timestep
// (runs before GEMM1's fused-stats atomics in stream order).
// ---------------------------------------------------------------------------
__global__ void tp_spike_input_kernel(const float* __restrict__ x,
                                      const float* __restrict__ delays,
                                      const float* __restrict__ beta_p,
                                      const float* __restrict__ thr_p,
                                      float* __restrict__ im,
                                      __bf16* __restrict__ spk,
                                      float* __restrict__ s1,
                                      float* __restrict__ s2,
                                      int t) {
    int idx = blockIdx.x * blockDim.x + threadIdx.x;   // B*IN threads
    if (idx < HIDDIM) { s1[idx] = 0.0f; s2[idx] = 0.0f; }
    float beta = beta_p[0];
    float thr  = thr_p[0];
    int b = idx >> 8;          // IN == 256
    int i = idx & (INDIM - 1);
    float xv = x[((size_t)b * TDIM + t) * INDIM + i];
    float v  = im[idx] * beta + xv * delays[i];
    float sp = (v - thr) > 0.0f ? 1.0f : 0.0f;
    im[idx]  = v * (1.0f - sp);
    spk[idx] = (__bf16)sp;
}

// ---------------------------------------------------------------------------
// TDM issue: 2D tile of bf16 weights -> LDS.
//   tile_dim0 = TP_KC (=64 elem = 128B rows), tile_dim1 = TP_BROWS (=128)
//   tensor_dim0_stride = K; pad: every 32 DWORDs (128B) insert 8 DWORDs (32B)
//   => LDS row pitch 160B.
// D# layout per CDNA5 ISA ch.8 (group0 128b, group1 256b).
// ---------------------------------------------------------------------------
#if defined(TP_HAVE_TDM)
typedef __attribute__((ext_vector_type(4))) unsigned int tp_u32x4;
typedef __attribute__((ext_vector_type(8))) int          tp_i32x8;
typedef __attribute__((ext_vector_type(4))) int          tp_i32x4;

__device__ __forceinline__ void tp_tdm_issue(const __bf16* gsrc, unsigned lds_off,
                                             int K, int Nrows) {
    unsigned long long ga = (unsigned long long)(uintptr_t)gsrc;
    tp_u32x4 g0;
    g0[0] = 1u;                                            // count=1, user mode
    g0[1] = lds_off;                                       // LDS byte address
    g0[2] = (unsigned)(ga & 0xFFFFFFFFu);                  // global_addr[31:0]
    g0[3] = (unsigned)((ga >> 32) & 0x1FFFFFFu) | (2u << 30); // [56:32] | type=2

    tp_i32x8 g1;
    //        data_size=2B   pad_en    interval=32dw  amount=8dw
    g1[0] = (int)((1u << 16) | (1u << 20) | (4u << 22) | (7u << 25));
    g1[1] = (int)(((unsigned)K & 0xFFFFu) << 16);          // tensor_dim0[15:0]
    g1[2] = (int)(((unsigned)K >> 16) |                    // tensor_dim0[31:16]
                  (((unsigned)Nrows & 0xFFFFu) << 16));    // tensor_dim1[15:0]
    g1[3] = (int)(((unsigned)Nrows >> 16) |                // tensor_dim1[31:16]
                  ((unsigned)TP_KC << 16));                // tile_dim0
    g1[4] = (int)((unsigned)TP_BROWS);                     // tile_dim1 (tile_dim2=0)
    g1[5] = (int)(unsigned)K;                              // tensor_dim0_stride[31:0]
    g1[6] = 0;                                             // stride hi / dim1_stride lo
    g1[7] = 0;

    tp_i32x4 gz = {0, 0, 0, 0};
#if __clang_major__ >= 23
    tp_i32x8 gz8 = {0, 0, 0, 0, 0, 0, 0, 0};
    __builtin_amdgcn_tensor_load_to_lds(g0, g1, gz, gz, gz8, 0);
#else
    __builtin_amdgcn_tensor_load_to_lds(g0, g1, gz, gz, 0);
#endif
}
#endif  // TP_HAVE_TDM

// ---------------------------------------------------------------------------
// WMMA bf16 GEMM + fused BN-stats epilogue:
//   C[M x N] = A[M x K] * W[N x K]^T   (f32 accumulate)
//   s1g[col] += sum_rows C, s2g[col] += sum_rows C^2  (block-local LDS
//   reduction via ds_add_f32, then one global atomicAdd per column).
// 256 threads = 8 waves (4 in M x 2 in N); wave tile 32(M) x 64(N).
// ---------------------------------------------------------------------------
__global__ __launch_bounds__(256)
void tp_gemm_bf16_wmma(const __bf16* __restrict__ A,
                       const __bf16* __restrict__ W,
                       float* __restrict__ C,
                       float* __restrict__ s1g,
                       float* __restrict__ s2g,
                       int M, int N, int K) {
    __shared__ __align__(32) char smem[2 * TP_BUFB];       // 40 KB staging
    __shared__ float redbuf[2 * TP_BROWS];                 // col sum / sumsq

    const int tid   = threadIdx.x;
    const int lane  = tid & 31;
    const int wid   = tid >> 5;          // 0..7
    const int waveM = wid & 3;           // 4 waves along M
    const int waveN = wid >> 2;          // 2 waves along N
    const int m0    = blockIdx.x * 128 + waveM * 32;
    const int nblk  = blockIdx.y * 128;  // block's first weight row (= C col)
    const int nw    = waveN * 64;        // wave's col offset inside block
    const int half  = lane >> 4;         // 0 or 1
    const int l16   = lane & 15;

    if (tid < 2 * TP_BROWS) redbuf[tid] = 0.0f;            // published by loop barriers

    v8f zero = {0.f, 0.f, 0.f, 0.f, 0.f, 0.f, 0.f, 0.f};
    v8f acc[2][4];
#pragma unroll
    for (int i = 0; i < 2; ++i)
#pragma unroll
        for (int j = 0; j < 4; ++j) acc[i][j] = zero;

    const __bf16* Arow = A + (size_t)(m0 + l16) * K;
    const unsigned lds0 = (unsigned)(uintptr_t)(&smem[0]); // LDS byte offset

    const int nch = K / TP_KC;

    // ---- stage chunk 0 ----
    {
        const __bf16* gsrc = W + (size_t)nblk * K;
#if defined(TP_HAVE_TDM)
        if (wid == 0) tp_tdm_issue(gsrc, lds0, K, N);
#else
        int row = tid >> 1, hr = tid & 1;
        const char* src = (const char*)(gsrc + (size_t)row * K) + hr * 64;
        char* dst = &smem[row * TP_ROWB + hr * 64];
#pragma unroll
        for (int i = 0; i < 4; ++i)
            *(uint4*)(dst + i * 16) = *(const uint4*)(src + i * 16);
#endif
    }

    for (int c = 0; c < nch; ++c) {
        // ---- stage chunk c+1 into the other buffer, wait for chunk c ----
        if (c + 1 < nch) {
            const __bf16* gsrc = W + (size_t)nblk * K + (size_t)(c + 1) * TP_KC;
            unsigned bufoff = (unsigned)((c + 1) & 1) * TP_BUFB;
#if defined(TP_HAVE_TDM)
            if (wid == 0) {
                tp_tdm_issue(gsrc, lds0 + bufoff, K, N);
                __builtin_amdgcn_s_wait_tensorcnt(1);      // chunk c done
            }
#else
            int row = tid >> 1, hr = tid & 1;
            const char* src = (const char*)(gsrc + (size_t)row * K) + hr * 64;
            char* dst = &smem[bufoff + row * TP_ROWB + hr * 64];
#pragma unroll
            for (int i = 0; i < 4; ++i)
                *(uint4*)(dst + i * 16) = *(const uint4*)(src + i * 16);
#endif
        } else {
#if defined(TP_HAVE_TDM)
            if (wid == 0) __builtin_amdgcn_s_wait_tensorcnt(0);
#endif
        }
        __syncthreads();                                   // LDS chunk c visible

        // ---- compute on chunk c ----
        const int cb = (c & 1) * TP_BUFB;
#pragma unroll
        for (int ks = 0; ks < TP_KC / 32; ++ks) {
            const int kg = c * TP_KC + ks * 32;            // global K base
            if (kg + 32 < K)
                __builtin_prefetch(Arow + kg + 32, 0, 1);  // global_prefetch_b8

            v16bf afrag[2];
#pragma unroll
            for (int fm = 0; fm < 2; ++fm) {
                // CDNA5 16-bit A(16x32): lane row m=l16, half selects K sets
                const __bf16* p = Arow + (size_t)fm * 16 * K + kg + half * 16;
                v8bf lo = *(const v8bf*)(p);               // K = kb+16h .. +7
                v8bf hi = *(const v8bf*)(p + 8);           // K = kb+16h+8 .. +7
                afrag[fm] = __builtin_shufflevector(lo, hi,
                    0, 1, 2, 3, 4, 5, 6, 7, 8, 9, 10, 11, 12, 13, 14, 15);
            }

            v16bf bfrag[4];
#pragma unroll
            for (int fn = 0; fn < 4; ++fn) {
                // B(32x16), contiguous k order as A: lane col n=l16, half 1 holds K 16..31
                int row = nw + fn * 16 + l16;              // weight row in tile
                const v16bf* p = (const v16bf*)
                    &smem[cb + row * TP_ROWB + ks * 64 + half * 32];
                bfrag[fn] = *p;                            // ds_load_b128 x2
            }

#pragma unroll
            for (int fm = 0; fm < 2; ++fm)
#pragma unroll
                for (int fn = 0; fn < 4; ++fn)
                    acc[fm][fn] = __builtin_amdgcn_wmma_f32_16x16x32_bf16(
                        false, afrag[fm], false, bfrag[fn],
                        (short)0, acc[fm][fn], false, false);
        }
        __syncthreads();                                   // buffer reusable
    }

    // ---- store C (lanes 0-15: rows r; lanes 16-31: rows r+8) ----
#pragma unroll
    for (int fm = 0; fm < 2; ++fm) {
#pragma unroll
        for (int fn = 0; fn < 4; ++fn) {
            int mrow = m0 + fm * 16 + half * 8;
            int col  = nblk + nw + fn * 16 + l16;
            float* cp = C + (size_t)mrow * N + col;
#pragma unroll
            for (int r = 0; r < 8; ++r)
                cp[(size_t)r * N] = acc[fm][fn][r];
        }
    }

    // ---- fused BN-stats epilogue -----------------------------------------
    // Each lane owns 16 rows (fm x r) of 4 columns (fn). Reduce rows in
    // registers, then across lanes/waves with LDS float atomics (ds_add_f32),
    // then one global atomicAdd per column for this block's 128 rows.
#pragma unroll
    for (int fn = 0; fn < 4; ++fn) {
        float cs = 0.0f, cq = 0.0f;
#pragma unroll
        for (int fm = 0; fm < 2; ++fm)
#pragma unroll
            for (int r = 0; r < 8; ++r) {
                float v = acc[fm][fn][r];
                cs += v;
                cq += v * v;
            }
        int col = nw + fn * 16 + l16;                      // 0..127 in block
        atomicAdd(&redbuf[col], cs);                       // ds_add_f32
        atomicAdd(&redbuf[TP_BROWS + col], cq);
    }
    __syncthreads();
    if (tid < TP_BROWS) {
        atomicAdd(&s1g[nblk + tid], redbuf[tid]);          // global_atomic_add_f32
        atomicAdd(&s2g[nblk + tid], redbuf[TP_BROWS + tid]);
    }
}

// ---------------------------------------------------------------------------
// BN apply (layer 1) + hidden membrane update + spike -> bf16 spikes.
// Also zeroes layer-2 BN stats accumulators for this timestep
// (runs before GEMM2's fused-stats atomics in stream order).
// ---------------------------------------------------------------------------
__global__ void tp_bn_hidden_kernel(const float* __restrict__ h,
                                    const float* __restrict__ s1,
                                    const float* __restrict__ s2,
                                    const float* __restrict__ g,
                                    const float* __restrict__ bb,
                                    const float* __restrict__ delays,
                                    const float* __restrict__ beta_p,
                                    const float* __restrict__ thr_p,
                                    float* __restrict__ hm,
                                    __bf16* __restrict__ spk2,
                                    float* __restrict__ s1b,
                                    float* __restrict__ s2b) {
    int idx = blockIdx.x * blockDim.x + threadIdx.x;   // B*HID threads
    if (idx < OUTDIM) { s1b[idx] = 0.0f; s2b[idx] = 0.0f; }
    int j = idx & (HIDDIM - 1);
    const float invB = 1.0f / (float)BDIM;
    float mu  = s1[j] * invB;
    float var = s2[j] * invB - mu * mu;
    float hn  = (h[idx] - mu) * rsqrtf(var + BN_EPS) * g[j] + bb[j];
    float beta = beta_p[0];
    float thr  = thr_p[0];
    float v  = hm[idx] * beta + hn * delays[j];
    float sp = (v - thr) > 0.0f ? 1.0f : 0.0f;
    hm[idx]  = v * (1.0f - sp);
    spk2[idx] = (__bf16)sp;
}

// ---------------------------------------------------------------------------
// BN apply (layer 2) + time-attention accumulate into the final output.
// ---------------------------------------------------------------------------
__global__ void tp_bn_out_kernel(const float* __restrict__ o,
                                 const float* __restrict__ s1b,
                                 const float* __restrict__ s2b,
                                 const float* __restrict__ g,
                                 const float* __restrict__ bb,
                                 const float* __restrict__ ta,
                                 float* __restrict__ out,
                                 int t) {
    int idx = blockIdx.x * blockDim.x + threadIdx.x;   // B*OUT threads
    int c = idx & (OUTDIM - 1);
    const float invB = 1.0f / (float)BDIM;
    float mu  = s1b[c] * invB;
    float var = s2b[c] * invB - mu * mu;
    float on  = (o[idx] - mu) * rsqrtf(var + BN_EPS) * g[c] + bb[c];
    out[idx] += ta[t] * on;
}

// ---------------------------------------------------------------------------
// Host-side orchestration (all launches on `stream`, graph-capture safe).
// ---------------------------------------------------------------------------
extern "C" void kernel_launch(void* const* d_in, const int* in_sizes, int n_in,
                              void* d_out, int out_size, void* d_ws, size_t ws_size,
                              hipStream_t stream) {
    (void)in_sizes; (void)n_in; (void)out_size; (void)ws_size;

    const float* x      = (const float*)d_in[0];   // [B, T, IN]
    const float* fc1_w  = (const float*)d_in[1];   // [HID, IN]
    // d_in[2] = fc1_b : cancelled exactly by training-mode BN mean subtraction
    const float* fc2_w  = (const float*)d_in[3];   // [OUT, HID]
    // d_in[4] = fc2_b : cancelled exactly by training-mode BN mean subtraction
    const float* bn1_g  = (const float*)d_in[5];
    const float* bn1_b  = (const float*)d_in[6];
    const float* bn2_g  = (const float*)d_in[7];
    const float* bn2_b  = (const float*)d_in[8];
    const float* cd     = (const float*)d_in[9];   // [1024]
    const float* lf     = (const float*)d_in[10];  // [1]
    // d_in[11] = surrogate_scale : only affects backward; forward-only here
    const float* in_thr = (const float*)d_in[12];
    const float* hd_thr = (const float*)d_in[13];
    const float* in_beta= (const float*)d_in[14];
    const float* hd_beta= (const float*)d_in[15];
    const float* ta     = (const float*)d_in[16];  // [T]

    float* out = (float*)d_out;                    // [B, OUT]

    // ---- workspace layout (256B-aligned regions) ----
    char* ws = (char*)d_ws;
    size_t off = 0;
    auto carve = [&](size_t bytes) -> char* {
        char* p = ws + off;
        off += (bytes + 255) & ~(size_t)255;
        return p;
    };
    float*  delays = (float*) carve(1024 * sizeof(float));
    __bf16* w1     = (__bf16*)carve((size_t)HIDDIM * INDIM * sizeof(__bf16));
    __bf16* w2     = (__bf16*)carve((size_t)OUTDIM * HIDDIM * sizeof(__bf16));
    float*  im     = (float*) carve((size_t)BDIM * INDIM * sizeof(float));
    float*  hm     = (float*) carve((size_t)BDIM * HIDDIM * sizeof(float));
    __bf16* spk1   = (__bf16*)carve((size_t)BDIM * INDIM * sizeof(__bf16));
    __bf16* spk2   = (__bf16*)carve((size_t)BDIM * HIDDIM * sizeof(__bf16));
    float*  hbuf   = (float*) carve((size_t)BDIM * HIDDIM * sizeof(float));
    float*  obuf   = (float*) carve((size_t)BDIM * OUTDIM * sizeof(float));
    float*  s1     = (float*) carve(HIDDIM * sizeof(float));
    float*  s2     = (float*) carve(HIDDIM * sizeof(float));
    float*  s1b    = (float*) carve(OUTDIM * sizeof(float));
    float*  s2b    = (float*) carve(OUTDIM * sizeof(float));

    // ---- init: zero output + membranes; prep delays + bf16 weights ----
    {
        int n;
        n = BDIM * OUTDIM;
        tp_zero_kernel<<<(n + 255) / 256, 256, 0, stream>>>(out, n);
        n = BDIM * INDIM;
        tp_zero_kernel<<<(n + 255) / 256, 256, 0, stream>>>(im, n);
        n = BDIM * HIDDIM;
        tp_zero_kernel<<<(n + 255) / 256, 256, 0, stream>>>(hm, n);

        int np = HIDDIM * INDIM;   // == OUT*HID == 262144, covers everything
        tp_prep_kernel<<<(np + 255) / 256, 256, 0, stream>>>(
            fc1_w, fc2_w, cd, lf, w1, w2, delays);
    }

    // ---- time loop: 5 launches per step, all on `stream` ----
    dim3 gemm1_grid(BDIM / 128, HIDDIM / 128);  // (8, 8)
    dim3 gemm2_grid(BDIM / 128, OUTDIM / 128);  // (8, 2)

    for (int t = 0; t < TDIM; ++t) {
        // zeroes s1/s2 before GEMM1's fused-stats atomics
        tp_spike_input_kernel<<<(BDIM * INDIM) / 256, 256, 0, stream>>>(
            x, delays, in_beta, in_thr, im, spk1, s1, s2, t);

        tp_gemm_bf16_wmma<<<gemm1_grid, 256, 0, stream>>>(
            spk1, w1, hbuf, s1, s2, BDIM, HIDDIM, INDIM);

        // zeroes s1b/s2b before GEMM2's fused-stats atomics
        tp_bn_hidden_kernel<<<(BDIM * HIDDIM) / 256, 256, 0, stream>>>(
            hbuf, s1, s2, bn1_g, bn1_b, delays, hd_beta, hd_thr,
            hm, spk2, s1b, s2b);

        tp_gemm_bf16_wmma<<<gemm2_grid, 256, 0, stream>>>(
            spk2, w2, obuf, s1b, s2b, BDIM, OUTDIM, HIDDIM);

        tp_bn_out_kernel<<<(BDIM * OUTDIM) / 256, 256, 0, stream>>>(
            obuf, s1b, s2b, bn2_g, bn2_b, ta, out, t);
    }
}
